// Block_21062519619681
// MI455X (gfx1250) — matrix operand order repaired and checked by reading, not compiled, not run on hardware
//
#include <hip/hip_runtime.h>
#include <math.h>

// ---------------- problem constants ----------------
constexpr int KB    = 4;
constexpr int KT    = 1024;
constexpr int KC    = 1024;
constexpr int KNH   = 16;
constexpr int KHD   = 64;
constexpr int KDS   = 64;
constexpr int KDIN  = 2048;   // EXP*C
constexpr int KNHM  = 32;     // DIN/PHD
constexpr int KPHD  = 64;
constexpr int KCONVD= 2176;   // DIN + 2*DS
constexpr int KNPROJ= 4256;   // 2*DIN + 2*DS + NHM
constexpr int KNATT = 1152;   // C + 2*HD
constexpr int KFFN  = 4096;
constexpr int KM    = KB * KT;  // 4096 tokens

typedef __attribute__((ext_vector_type(16))) __bf16 v16bf;
typedef __attribute__((ext_vector_type(8)))  float  v8f;

__device__ __forceinline__ unsigned short f2bf(float f) {
    unsigned u = __float_as_uint(f);
    unsigned r = u + 0x7fffu + ((u >> 16) & 1u);   // round-to-nearest-even
    return (unsigned short)(r >> 16);
}
__device__ __forceinline__ float sigm(float x) { return 1.f / (1.f + expf(-x)); }

// ---------------- weight transpose + f32->bf16, zero-padded rows ----------------
// W is K x N row-major (f32); Wt is Npad x K row-major (bf16), rows [N,Npad) = 0
__global__ __launch_bounds__(256) void wcvt_t_k(const float* __restrict__ W,
                                                unsigned short* __restrict__ Wt,
                                                int K, int N, int Npad) {
    size_t i = (size_t)blockIdx.x * 256 + threadIdx.x;
    if (i >= (size_t)Npad * K) return;
    int n = (int)(i / K);
    int k = (int)(i % K);
    Wt[i] = (n < N) ? f2bf(W[(size_t)k * N + n]) : (unsigned short)0;
}

// ---------------- rmsnorm over last dim ----------------
__global__ __launch_bounds__(256) void rmsnorm_k(const float* __restrict__ x,
                                                 unsigned short* __restrict__ obf,
                                                 float* __restrict__ of, int C) {
    int row = blockIdx.x;
    const float* xr = x + (size_t)row * C;
    __shared__ float red[256];
    float ss = 0.f;
    for (int j = threadIdx.x; j < C; j += 256) { float v = xr[j]; ss += v * v; }
    red[threadIdx.x] = ss; __syncthreads();
    for (int s = 128; s > 0; s >>= 1) {
        if (threadIdx.x < s) red[threadIdx.x] += red[threadIdx.x + s];
        __syncthreads();
    }
    float scale = rsqrtf(red[0] / (float)C + 1e-5f);
    for (int j = threadIdx.x; j < C; j += 256) {
        float v = xr[j] * scale;
        if (obf) obf[(size_t)row * C + j] = f2bf(v);
        if (of)  of[(size_t)row * C + j] = v;
    }
}

// ---------------- bf16 WMMA GEMM:  C[MxN] = A[MxK] * Bt[NpadxK]^T ----------------
// 128 threads = 4 waves. Block tile: 128(M) x 64(N); each wave 32(M) x 64(N).
// B tile (64 rows x 32 K, bf16) staged in LDS via async-to-LDS DMA, double buffered.
__global__ __launch_bounds__(128) void gemm_bf16_k(const unsigned short* __restrict__ A,
                                                   const unsigned short* __restrict__ Bt,
                                                   float* __restrict__ C,
                                                   int M, int N, int K) {
    __shared__ alignas(16) unsigned short bsm[2][64 * 32];
    int tid  = threadIdx.x;
    int lane = tid & 31, wave = tid >> 5;
    int r = lane & 15, g = lane >> 4;
    int mt = blockIdx.y * 128 + wave * 32;
    int nt = blockIdx.x * 64;

    // each thread DMAs 32 bytes (16 bf16) of the 4KB B tile per K-step
    int brow  = tid >> 1;           // 0..63
    int bhalf = (tid & 1) << 4;     // element 0 or 16 within the 32-wide K slice
    const unsigned short* bsrc = Bt + (size_t)(nt + brow) * K + bhalf;
    unsigned ldsA = (unsigned)(size_t)&bsm[0][brow * 32 + bhalf];
    unsigned ldsB = (unsigned)(size_t)&bsm[1][brow * 32 + bhalf];

    v8f acc[8];   // [mhalf(2)][j(4)]
#pragma unroll
    for (int j = 0; j < 8; ++j)
#pragma unroll
        for (int i = 0; i < 8; ++i) acc[j][i] = 0.f;

    const unsigned short* arow0 = A + (size_t)(mt + r) * K;
    const unsigned short* arow1 = A + (size_t)(mt + 16 + r) * K;

    // preload tile 0 into buffer 0 (ASYNCcnt-tracked LDS DMA)
    asm volatile("global_load_async_to_lds_b128 %0, %1, off"
                 :: "v"(ldsA), "v"(bsrc) : "memory");
    asm volatile("global_load_async_to_lds_b128 %0, %1, off"
                 :: "v"(ldsA + 16), "v"(bsrc + 8) : "memory");

    int nk = K >> 5;
    for (int kk = 0; kk < nk; ++kk) {
        asm volatile("s_wait_asynccnt 0x0" ::: "memory");
        __syncthreads();
        // stage next tile into the other buffer while computing this one
        if (kk + 1 < nk) {
            const unsigned short* src = bsrc + (size_t)(kk + 1) * 32;
            unsigned dst = (kk & 1) ? ldsA : ldsB;
            asm volatile("global_load_async_to_lds_b128 %0, %1, off"
                         :: "v"(dst), "v"(src) : "memory");
            asm volatile("global_load_async_to_lds_b128 %0, %1, off"
                         :: "v"(dst + 16), "v"(src + 8) : "memory");
        }
        int k0 = kk << 5;
        // A frags: lanes 0-15 hold K 0-7/16-23, lanes 16-31 hold K 8-15/24-31
        alignas(32) uint4 aq[2];
        aq[0] = *(const uint4*)(arow0 + k0 + g * 8);
        aq[1] = *(const uint4*)(arow0 + k0 + 16 + g * 8);
        v16bf af0 = *(const v16bf*)aq;
        aq[0] = *(const uint4*)(arow1 + k0 + g * 8);
        aq[1] = *(const uint4*)(arow1 + k0 + 16 + g * 8);
        v16bf af1 = *(const v16bf*)aq;
        const unsigned short* bcur = &bsm[kk & 1][0];
#pragma unroll
        for (int j = 0; j < 4; ++j) {
            // B frag: col = lane&15 (row n of Bt), lane half g holds K g*8..g*8+7 / 16+g*8..16+g*8+7
            const unsigned short* bp = bcur + (j * 16 + r) * 32 + g * 8;
            alignas(32) uint4 bq[2];
            bq[0] = *(const uint4*)bp;
            bq[1] = *(const uint4*)(bp + 16);
            v16bf bf = *(const v16bf*)bq;
            acc[j]     = __builtin_amdgcn_wmma_f32_16x16x32_bf16(
                false, af0, false, bf, (short)0, acc[j],     false, false);
            acc[4 + j] = __builtin_amdgcn_wmma_f32_16x16x32_bf16(
                false, af1, false, bf, (short)0, acc[4 + j], false, false);
        }
    }
    // C/D layout: vgpr i -> M = g*8 + i, N = lane&15
#pragma unroll
    for (int j = 0; j < 4; ++j) {
        int n = nt + j * 16 + r;
        if (n < N) {
#pragma unroll
            for (int i = 0; i < 8; ++i) {
                C[(size_t)(mt + (g << 3) + i) * N + n]      = acc[j][i];
                C[(size_t)(mt + 16 + (g << 3) + i) * N + n] = acc[4 + j][i];
            }
        }
    }
}

// ---------------- depthwise causal conv (width 4) + bias + silu ----------------
__global__ __launch_bounds__(256) void conv_silu_k(const float* __restrict__ zx,
                                                   const float* __restrict__ cw,
                                                   const float* __restrict__ cb,
                                                   float* __restrict__ xc) {
    size_t i = (size_t)blockIdx.x * 256 + threadIdx.x;
    if (i >= (size_t)KM * KCONVD) return;
    int c = (int)(i % KCONVD);
    size_t m = i / KCONVD;
    int t = (int)(m & (KT - 1));
    float acc = cb[c];
#pragma unroll
    for (int j = 0; j < 4; ++j) {
        int tt = t - 3 + j;
        if (tt >= 0) acc += zx[(m - 3 + j) * KNPROJ + KDIN + c] * cw[c * 4 + j];
    }
    xc[i] = acc * sigm(acc);
}

// ---------------- mamba2 selective scan: one block per (b, head) ----------------
__global__ __launch_bounds__(64) void mamba_scan_k(const float* __restrict__ zx,
                                                   const float* __restrict__ xc,
                                                   const float* __restrict__ dt_bias,
                                                   const float* __restrict__ A_log,
                                                   const float* __restrict__ Dp,
                                                   float* __restrict__ ybuf) {
    int bh = blockIdx.x;
    int b = bh >> 5, h = bh & 31;
    int p = threadIdx.x;                  // 0..63 (PHD)
    __shared__ float sB[KDS], sC[KDS], sSc[2];
    float hs[KDS];
#pragma unroll
    for (int n = 0; n < KDS; ++n) hs[n] = 0.f;
    float A = -expf(A_log[h]);
    float bias = dt_bias[h];
    float Dh = Dp[h];
    for (int t = 0; t < KT; ++t) {
        size_t row = (size_t)(b * KT + t);
        size_t rz = row * KNPROJ, rc = row * KCONVD;
        if (p == 0) {
            float raw = zx[rz + (KNPROJ - KNHM) + h] + bias;
            float dt = raw > 20.f ? raw : log1pf(expf(raw));   // softplus
            sSc[0] = dt;
            sSc[1] = expf(dt * A);
        }
        sB[p] = xc[rc + KDIN + p];
        sC[p] = xc[rc + KDIN + KDS + p];
        float xs = xc[rc + h * KPHD + p];
        __syncthreads();
        float dt = sSc[0], dA = sSc[1];
        float dtx = dt * xs;
        float y = 0.f;
#pragma unroll
        for (int n = 0; n < KDS; ++n) {
            hs[n] = dA * hs[n] + dtx * sB[n];
            y += hs[n] * sC[n];
        }
        ybuf[row * KDIN + h * KPHD + p] = y + Dh * xs;
        __syncthreads();
    }
}

// ---------------- g = y*silu(z), rmsnorm * mnorm_w -> bf16 ----------------
__global__ __launch_bounds__(256) void gate_norm_k(const float* __restrict__ zx,
                                                   const float* __restrict__ ybuf,
                                                   const float* __restrict__ mw,
                                                   unsigned short* __restrict__ gout) {
    int row = blockIdx.x;
    __shared__ float gs[KDIN];
    __shared__ float red[256];
    size_t rz = (size_t)row * KNPROJ, ry = (size_t)row * KDIN;
    float ss = 0.f;
    for (int j = threadIdx.x; j < KDIN; j += 256) {
        float z = zx[rz + j];
        float g = ybuf[ry + j] * (z * sigm(z));
        gs[j] = g; ss += g * g;
    }
    red[threadIdx.x] = ss; __syncthreads();
    for (int s = 128; s > 0; s >>= 1) {
        if (threadIdx.x < s) red[threadIdx.x] += red[threadIdx.x + s];
        __syncthreads();
    }
    float scale = rsqrtf(red[0] / (float)KDIN + 1e-5f);
    for (int j = threadIdx.x; j < KDIN; j += 256)
        gout[ry + j] = f2bf(gs[j] * scale * mw[j]);
}

// ---------------- causal MQA attention, one block per (b,h,t) row ----------------
__global__ __launch_bounds__(256) void mqa_attn_k(const float* __restrict__ qkv,
                                                  unsigned short* __restrict__ yout) {
    int t = blockIdx.x, h = blockIdx.y, b = blockIdx.z;
    int tid = threadIdx.x;
    __shared__ float sc[KT];
    __shared__ float qs[KHD];
    __shared__ float red[256];
    __shared__ float acc4[4][KHD];
    size_t rowq = (size_t)(b * KT + t) * KNATT;
    if (tid < KHD) qs[tid] = qkv[rowq + h * KHD + tid];
    __syncthreads();
    float lm = -3.0e38f;
    for (int s = tid; s <= t; s += 256) {
        const float* kr = qkv + (size_t)(b * KT + s) * KNATT + KC;
        float d = 0.f;
#pragma unroll
        for (int i = 0; i < KHD; ++i) d += qs[i] * kr[i];
        d *= 0.125f;
        sc[s] = d;
        lm = fmaxf(lm, d);
    }
    red[tid] = lm; __syncthreads();
    for (int s = 128; s > 0; s >>= 1) {
        if (tid < s) red[tid] = fmaxf(red[tid], red[tid + s]);
        __syncthreads();
    }
    float mx = red[0]; __syncthreads();
    float ls = 0.f;
    for (int s = tid; s <= t; s += 256) {
        float e = expf(sc[s] - mx);
        sc[s] = e; ls += e;
    }
    red[tid] = ls; __syncthreads();
    for (int s = 128; s > 0; s >>= 1) {
        if (tid < s) red[tid] += red[tid + s];
        __syncthreads();
    }
    float inv = 1.f / red[0];
    int d = tid & 63, qd = tid >> 6;
    float a = 0.f;
    for (int s = qd; s <= t; s += 4)
        a += sc[s] * qkv[(size_t)(b * KT + s) * KNATT + (KC + KHD) + d];
    acc4[qd][d] = a;
    __syncthreads();
    if (tid < KHD) {
        float y = (acc4[0][d] + acc4[1][d] + acc4[2][d] + acc4[3][d]) * inv;
        yout[(size_t)(b * KT + t) * KC + h * KHD + d] = f2bf(y);
    }
}

// ---------------- residual add ----------------
__global__ __launch_bounds__(256) void add2_k(float* __restrict__ dst,
                                              const float* __restrict__ a,
                                              const float* __restrict__ b, size_t n) {
    size_t i = (size_t)blockIdx.x * 256 + threadIdx.x;
    if (i < n) dst[i] = a[i] + b[i];
}

// ---------------- cmix token-shift mixing -> bf16 xk, xr ----------------
__global__ __launch_bounds__(256) void mix_k(const float* __restrict__ xn,
                                             const float* __restrict__ mk,
                                             const float* __restrict__ mr,
                                             unsigned short* __restrict__ xk,
                                             unsigned short* __restrict__ xr) {
    size_t i = (size_t)blockIdx.x * 256 + threadIdx.x;
    if (i >= (size_t)KM * KC) return;
    int c = (int)(i & (KC - 1));
    size_t m = i >> 10;
    int t = (int)(m & (KT - 1));
    float cur = xn[i];
    float prev = (t > 0) ? xn[i - KC] : 0.f;
    float xx = prev - cur;
    xk[i] = f2bf(cur + xx * mk[c]);
    xr[i] = f2bf(cur + xx * mr[c]);
}

// ---------------- 0.5*(1+erf((k-mu)/denom)) -> bf16 ----------------
__global__ __launch_bounds__(256) void erf_act_k(const float* __restrict__ kin,
                                                 unsigned short* __restrict__ kout) {
    size_t i = (size_t)blockIdx.x * 256 + threadIdx.x;
    if (i >= (size_t)KM * KFFN) return;
    float v = kin[i];
    // mu = sqrt(0.5), denom = sqrt(1/(4pi))*sqrt(2); 1/denom = 2.50662827
    kout[i] = f2bf(0.5f * (1.f + erff((v - 0.70710678f) * 2.50662827f)));
}

// ---------------- out = xcur + sigmoid(r) * kv ----------------
__global__ __launch_bounds__(256) void final_k(const float* __restrict__ xcur,
                                               const float* __restrict__ rbuf,
                                               const float* __restrict__ kv,
                                               float* __restrict__ out, size_t n) {
    size_t i = (size_t)blockIdx.x * 256 + threadIdx.x;
    if (i < n) out[i] = xcur[i] + sigm(rbuf[i]) * kv[i];
}

// =====================================================================
extern "C" void kernel_launch(void* const* d_in, const int* in_sizes, int n_in,
                              void* d_out, int out_size, void* d_ws, size_t ws_size,
                              hipStream_t stream) {
    (void)in_sizes; (void)n_in; (void)out_size; (void)ws_size;
    const float* x        = (const float*)d_in[0];
    const float* in_proj  = (const float*)d_in[1];
    const float* conv_w   = (const float*)d_in[2];
    const float* conv_b   = (const float*)d_in[3];
    const float* dt_bias  = (const float*)d_in[4];
    const float* A_log    = (const float*)d_in[5];
    const float* Dp       = (const float*)d_in[6];
    const float* mnorm_w  = (const float*)d_in[7];
    const float* out_proj = (const float*)d_in[8];
    const float* attn_w   = (const float*)d_in[9];
    const float* proj_w   = (const float*)d_in[10];
    const float* maa_k    = (const float*)d_in[11];
    const float* maa_r    = (const float*)d_in[12];
    const float* key_w    = (const float*)d_in[13];
    const float* recept_w = (const float*)d_in[14];
    const float* value_w  = (const float*)d_in[15];
    float* out = (float*)d_out;

    auto pad64 = [](int n) { return (n + 63) & ~63; };
    const int NPROJ_P = pad64(KNPROJ);   // 4288

    // -------- workspace carve-up (256B aligned) --------
    char* p = (char*)d_ws;
    auto alloc = [&](size_t bytes) -> char* {
        char* r = p; p += (bytes + 255) & ~(size_t)255; return r;
    };
    unsigned short* w_in_t  = (unsigned short*)alloc((size_t)NPROJ_P * KC * 2);
    unsigned short* w_out_t = (unsigned short*)alloc((size_t)KC * KDIN * 2);
    unsigned short* w_att_t = (unsigned short*)alloc((size_t)KNATT * KC * 2);
    unsigned short* w_prj_t = (unsigned short*)alloc((size_t)KC * KC * 2);
    unsigned short* w_key_t = (unsigned short*)alloc((size_t)KFFN * KC * 2);
    unsigned short* w_rec_t = (unsigned short*)alloc((size_t)KC * KC * 2);
    unsigned short* w_val_t = (unsigned short*)alloc((size_t)KC * KFFN * 2);
    unsigned short* a_bf    = (unsigned short*)alloc((size_t)KM * KC * 2);   // xn / yatt
    float* zx               = (float*)alloc((size_t)KM * KNPROJ * 4);        // also kbuf
    float* xconv            = (float*)alloc((size_t)KM * KCONVD * 4);        // also qkv / kact
    float* ybuf             = (float*)alloc((size_t)KM * KDIN * 4);          // also xn3+xk+xr
    unsigned short* g_bf    = (unsigned short*)alloc((size_t)KM * KDIN * 2);
    float* tmp1             = (float*)alloc((size_t)KM * KC * 4);
    float* xcur             = (float*)alloc((size_t)KM * KC * 4);
    float* rbuf             = (float*)alloc((size_t)KM * KC * 4);

    float* qkv              = xconv;                                  // stage-2 alias
    float* kbuf             = zx;                                     // stage-3 alias
    unsigned short* kact_bf = (unsigned short*)xconv;                 // stage-3 alias
    float* xn3              = ybuf;                                   // stage-3 alias
    unsigned short* xk_bf   = (unsigned short*)((char*)ybuf + (size_t)KM * KC * 4);
    unsigned short* xr_bf   = (unsigned short*)((char*)ybuf + (size_t)KM * KC * 4
                                                            + (size_t)KM * KC * 2);

    auto cdiv = [](size_t a, size_t b) -> unsigned { return (unsigned)((a + b - 1) / b); };
    const size_t nMC = (size_t)KM * KC;

    // -------- weights -> bf16 transposed (rows zero-padded to 64) --------
    wcvt_t_k<<<cdiv((size_t)NPROJ_P * KC, 256), 256, 0, stream>>>(in_proj,  w_in_t,  KC,   KNPROJ, NPROJ_P);
    wcvt_t_k<<<cdiv((size_t)KC * KDIN, 256),    256, 0, stream>>>(out_proj, w_out_t, KDIN, KC,     KC);
    wcvt_t_k<<<cdiv((size_t)KNATT * KC, 256),   256, 0, stream>>>(attn_w,   w_att_t, KC,   KNATT,  KNATT);
    wcvt_t_k<<<cdiv((size_t)KC * KC, 256),      256, 0, stream>>>(proj_w,   w_prj_t, KC,   KC,     KC);
    wcvt_t_k<<<cdiv((size_t)KFFN * KC, 256),    256, 0, stream>>>(key_w,    w_key_t, KC,   KFFN,   KFFN);
    wcvt_t_k<<<cdiv((size_t)KC * KC, 256),      256, 0, stream>>>(recept_w, w_rec_t, KC,   KC,     KC);
    wcvt_t_k<<<cdiv((size_t)KC * KFFN, 256),    256, 0, stream>>>(value_w,  w_val_t, KFFN, KC,     KC);

    // ================= stage 1: mamba2 =================
    rmsnorm_k<<<KM, 256, 0, stream>>>(x, a_bf, nullptr, KC);
    gemm_bf16_k<<<dim3(cdiv(KNPROJ, 64), KM / 128), 128, 0, stream>>>(a_bf, w_in_t, zx, KM, KNPROJ, KC);
    conv_silu_k<<<cdiv((size_t)KM * KCONVD, 256), 256, 0, stream>>>(zx, conv_w, conv_b, xconv);
    mamba_scan_k<<<KB * KNHM, 64, 0, stream>>>(zx, xconv, dt_bias, A_log, Dp, ybuf);
    gate_norm_k<<<KM, 256, 0, stream>>>(zx, ybuf, mnorm_w, g_bf);
    gemm_bf16_k<<<dim3(cdiv(KC, 64), KM / 128), 128, 0, stream>>>(g_bf, w_out_t, tmp1, KM, KC, KDIN);
    add2_k<<<cdiv(nMC, 256), 256, 0, stream>>>(xcur, x, tmp1, nMC);

    // ================= stage 2: MQA =================
    rmsnorm_k<<<KM, 256, 0, stream>>>(xcur, a_bf, nullptr, KC);
    gemm_bf16_k<<<dim3(cdiv(KNATT, 64), KM / 128), 128, 0, stream>>>(a_bf, w_att_t, qkv, KM, KNATT, KC);
    mqa_attn_k<<<dim3(KT, KNH, KB), 256, 0, stream>>>(qkv, a_bf);
    gemm_bf16_k<<<dim3(cdiv(KC, 64), KM / 128), 128, 0, stream>>>(a_bf, w_prj_t, tmp1, KM, KC, KC);
    add2_k<<<cdiv(nMC, 256), 256, 0, stream>>>(xcur, xcur, tmp1, nMC);

    // ================= stage 3: cmix =================
    rmsnorm_k<<<KM, 256, 0, stream>>>(xcur, nullptr, xn3, KC);
    mix_k<<<cdiv(nMC, 256), 256, 0, stream>>>(xn3, maa_k, maa_r, xk_bf, xr_bf);
    gemm_bf16_k<<<dim3(cdiv(KFFN, 64), KM / 128), 128, 0, stream>>>(xk_bf, w_key_t, kbuf, KM, KFFN, KC);
    gemm_bf16_k<<<dim3(cdiv(KC, 64), KM / 128), 128, 0, stream>>>(xr_bf, w_rec_t, rbuf, KM, KC, KC);
    erf_act_k<<<cdiv((size_t)KM * KFFN, 256), 256, 0, stream>>>(kbuf, kact_bf);
    gemm_bf16_k<<<dim3(cdiv(KC, 64), KM / 128), 128, 0, stream>>>(kact_bf, w_val_t, tmp1, KM, KC, KFFN);
    final_k<<<cdiv(nMC, 256), 256, 0, stream>>>(xcur, rbuf, tmp1, out, nMC);
}
